// DNAEncoder_33328946216993
// MI455X (gfx1250) — matrix operand order repaired and checked by reading, not compiled, not run on hardware
//
#include <hip/hip_runtime.h>
#include <hip/hip_bf16.h>

// ---------------------------------------------------------------------------
// DNA transformer encoder on gfx1250 (MI455X).
//  - all matmuls on V_WMMA_F32_16X16X32_BF16 (bf16 operands, f32 accumulate)
//  - down-projection GEMM stages A tiles via the Tensor Data Mover
//    (tensor_load_to_lds + s_wait_tensorcnt), double buffered in LDS
// ---------------------------------------------------------------------------

typedef __bf16 bf16_t;
typedef __attribute__((ext_vector_type(16))) __bf16        v16bf;
typedef __attribute__((ext_vector_type(8)))  float         v8f;
typedef __attribute__((ext_vector_type(4)))  unsigned int  u32x4;
typedef __attribute__((ext_vector_type(8)))  unsigned int  u32x8;

#define Bz      4
#define Lz      1024
#define Dz      512
#define Hz      8
#define DEPTHz  6
#define PROJz   128
#define HIDz    2048
#define HDz     64
#define Sz      1025          // L + 1 (cls)
#define SPz     1056          // S padded to multiple of 32
#define MTOT    (Bz * SPz)    // 4224 = 264 row-tiles = 66 blocks * 4 waves
#define EPSz    1e-5f

// TDM staging for the down-projection GEMM
#define TDK     64            // K-chunk (elements)
#define LDSLD   72            // LDS row stride in elements (64 + 4-DWORD pad)

// ---------------------------------------------------------------------------
// WMMA fragment helpers (wave32; A and B both use contiguous k = 16*half + j)
// ---------------------------------------------------------------------------

__device__ __forceinline__ v8f wmma_bf16(v16bf a, v16bf b, v8f c) {
  return __builtin_amdgcn_wmma_f32_16x16x32_bf16(false, a, false, b, (short)0, c,
                                                 false, false);
}

// A-matrix 16x32 bf16 tile; lane l: m=l&15, half=l>>4; elem j -> K=k0+16*half+j.
__device__ __forceinline__ v16bf load_a16(const bf16_t* __restrict__ Arow,
                                          int lda, int k0, int lane) {
  const int m  = lane & 15;
  const int kb = (lane >> 4) << 4;
  const bf16_t* p = Arow + (size_t)m * lda + k0 + kb;
  v16bf a;
#pragma unroll
  for (int j = 0; j < 8; ++j) a[j] = p[j];
#pragma unroll
  for (int j = 0; j < 8; ++j) a[8 + j] = p[8 + j];
  return a;
}

// B-matrix 32x16 bf16 (KxN) from N-major storage: elem (k,n)=Bt[n*ldb+k].
__device__ __forceinline__ v16bf load_b16(const bf16_t* __restrict__ Bt,
                                          int ldb, int k0, int lane) {
  const int n  = lane & 15;
  const int kb = (lane >> 4) << 4;
  const bf16_t* p = Bt + (size_t)n * ldb + k0 + kb;
  v16bf b;
#pragma unroll
  for (int j = 0; j < 16; ++j) b[j] = p[j];
  return b;
}

// B-matrix 32x16 where elem (k,n) = Bp[k*rowStride + n] (K-major, e.g. V).
__device__ __forceinline__ v16bf load_b16_strided(const bf16_t* __restrict__ Bp,
                                                  int rowStride, int lane) {
  const int n  = lane & 15;
  const int kb = (lane >> 4) << 4;
  v16bf b;
#pragma unroll
  for (int j = 0; j < 16; ++j) b[j] = Bp[(size_t)(kb + j) * rowStride + n];
  return b;
}

__device__ __forceinline__ float gelu_f(float x) {
  return 0.5f * x * (1.0f + erff(x * 0.70710678118654752f));
}

__device__ __forceinline__ v8f zero8() {
  v8f z = {0.f, 0.f, 0.f, 0.f, 0.f, 0.f, 0.f, 0.f};
  return z;
}

// ---------------------------------------------------------------------------
// Tensor Data Mover: load a 2D bf16 tile (tile_d0 x tile_d1, row stride
// stride_elems) from global ga into LDS at lds_addr, with LDS padding of
// 4 DWORDs after every 32 DWORDs (row stride 64 -> 72 elements).
// D# layout per CDNA5 ISA. Groups 2/3 zeroed
// (2D tile: tile_dim2/3 = 0 -> unused).
// ---------------------------------------------------------------------------
__device__ __forceinline__ void tdm_load_tile(unsigned lds_addr,
                                              unsigned long long ga,
                                              unsigned tile_d0, unsigned tile_d1,
                                              unsigned long long stride_elems) {
  const unsigned td0 = 0x7FFFFFFFu, td1 = 0x7FFFFFFFu;  // no OOB clipping
  u32x4 g0;
  g0[0] = 1u;                                               // count=1 (valid)
  g0[1] = lds_addr;                                         // LDS byte address
  g0[2] = (unsigned)ga;                                     // global_addr[31:0]
  g0[3] = (unsigned)((ga >> 32) & 0x01FFFFFFull) | (2u << 30); // [56:32] | type=2
  u32x8 g1;
  // data_size=1 (2B) | pad_enable | pad_interval=4 (32 DW) | pad_amount=3 (4 DW)
  g1[0] = 0x00010000u | (1u << 20) | (4u << 22) | (3u << 25);
  g1[1] = (td0 & 0xFFFFu) << 16;                            // abar=0, td0 lo16
  g1[2] = (td0 >> 16) | ((td1 & 0xFFFFu) << 16);            // td0 hi16, td1 lo16
  g1[3] = (td1 >> 16) | (tile_d0 << 16);                    // td1 hi16, tile_dim0
  g1[4] = tile_d1 & 0xFFFFu;                                // tile_dim1, tile_dim2=0
  g1[5] = (unsigned)(stride_elems & 0xFFFFFFFFull);         // dim0 stride lo32
  g1[6] = (unsigned)((stride_elems >> 32) & 0xFFFFull);     // dim0 stride hi16
  g1[7] = 0u;
  u32x4 gz = {0u, 0u, 0u, 0u};
  asm volatile("tensor_load_to_lds %0, %1, %2, %3"
               :: "s"(g0), "s"(g1), "s"(gz), "s"(gz)
               : "memory");
}

// ---------------------------------------------------------------------------
// Weight transpose + cast: W[z][K][N] f32 -> Wt[z][N][K] bf16
// ---------------------------------------------------------------------------
__global__ void tcast_kernel(const float* __restrict__ W, bf16_t* __restrict__ Wt,
                             int K, int N) {
  const size_t off = (size_t)blockIdx.z * K * N;
  const int idx = blockIdx.x * blockDim.x + threadIdx.x;
  if (idx >= K * N) return;
  const int kk = idx / N, nn = idx % N;
  Wt[off + (size_t)nn * K + kk] = (bf16_t)W[off + idx];
}

// ---------------------------------------------------------------------------
// Token embedding + cls + zero padding: h[MTOT][D] f32
// ---------------------------------------------------------------------------
__global__ void embed_kernel(const int* __restrict__ x, const float* __restrict__ emb,
                             const float* __restrict__ cls, float* __restrict__ h) {
  const int row = blockIdx.x;
  const int b = row / SPz, s = row % SPz;
  const int t = threadIdx.x;
  float* hr = h + (size_t)row * Dz;
  if (s == 0) {
    hr[t] = cls[t]; hr[t + 256] = cls[t + 256];
  } else if (s < Sz) {
    const float* e = emb + (size_t)x[b * Lz + s - 1] * Dz;
    hr[t] = e[t]; hr[t + 256] = e[t + 256];
  } else {
    hr[t] = 0.f; hr[t + 256] = 0.f;
  }
}

__global__ void mask_kernel(const int* __restrict__ x, float* __restrict__ km) {
  const int idx = blockIdx.x * blockDim.x + threadIdx.x;
  if (idx >= Bz * SPz) return;
  const int b = idx / SPz, j = idx % SPz;
  float m = -1e30f;
  if (j == 0) m = 0.f;
  else if (j < Sz && x[b * Lz + j - 1] != 0) m = 0.f;
  km[idx] = m;
}

// ---------------------------------------------------------------------------
// LayerNorm over D=512; 256 threads/row, outputs f32 (optional) and bf16
// ---------------------------------------------------------------------------
__global__ void ln_kernel(const float* __restrict__ x, const float* __restrict__ g,
                          const float* __restrict__ b, float* __restrict__ outf,
                          bf16_t* __restrict__ outb) {
  __shared__ float redS[8];
  __shared__ float redQ[8];
  const int row = blockIdx.x;
  const int t = threadIdx.x;
  const float* xr = x + (size_t)row * Dz;
  const float v0 = xr[t], v1 = xr[t + 256];
  float s = v0 + v1;
  float q = v0 * v0 + v1 * v1;
#pragma unroll
  for (int off = 16; off; off >>= 1) {
    s += __shfl_xor(s, off, 32);
    q += __shfl_xor(q, off, 32);
  }
  const int wv = t >> 5, ln = t & 31;
  if (ln == 0) { redS[wv] = s; redQ[wv] = q; }
  __syncthreads();
  float mean = 0.f, msq = 0.f;
#pragma unroll
  for (int i = 0; i < 8; ++i) { mean += redS[i]; msq += redQ[i]; }
  mean *= (1.f / Dz); msq *= (1.f / Dz);
  const float inv = rsqrtf(msq - mean * mean + EPSz);
  const float o0 = (v0 - mean) * inv * g[t] + b[t];
  const float o1 = (v1 - mean) * inv * g[t + 256] + b[t + 256];
  if (outf) {
    outf[(size_t)row * Dz + t] = o0;
    outf[(size_t)row * Dz + t + 256] = o1;
  }
  outb[(size_t)row * Dz + t] = (bf16_t)o0;
  outb[(size_t)row * Dz + t + 256] = (bf16_t)o1;
}

// ---------------------------------------------------------------------------
// Fused Q/K/V projection: one A-fragment feeds three WMMA accumulator sets.
// grid = (D/16, MTOT/64), block = 128 (4 waves). Exact grid -> EXEC all-ones.
// ---------------------------------------------------------------------------
__global__ void qkv_kernel(const bf16_t* __restrict__ A,
                           const bf16_t* __restrict__ qWt, const bf16_t* __restrict__ kWt,
                           const bf16_t* __restrict__ vWt,
                           const float* __restrict__ qb, const float* __restrict__ kb,
                           const float* __restrict__ vb,
                           bf16_t* __restrict__ qo, bf16_t* __restrict__ ko,
                           bf16_t* __restrict__ vo) {
  const int wave = threadIdx.x >> 5, lane = threadIdx.x & 31;
  const int ntile = blockIdx.x;
  const int mtile = blockIdx.y * 4 + wave;
  const bf16_t* Arow = A + (size_t)mtile * 16 * Dz;
  const bf16_t* Bq = qWt + (size_t)ntile * 16 * Dz;
  const bf16_t* Bk = kWt + (size_t)ntile * 16 * Dz;
  const bf16_t* Bv = vWt + (size_t)ntile * 16 * Dz;
  v8f aq = zero8(), ak = zero8(), av = zero8();
  for (int k0 = 0; k0 < Dz; k0 += 32) {
    const v16bf a = load_a16(Arow, Dz, k0, lane);
    aq = wmma_bf16(a, load_b16(Bq, Dz, k0, lane), aq);
    ak = wmma_bf16(a, load_b16(Bk, Dz, k0, lane), ak);
    av = wmma_bf16(a, load_b16(Bv, Dz, k0, lane), av);
  }
  const int n = ntile * 16 + (lane & 15);
  const float bq = qb[n], bk = kb[n], bv = vb[n];
  const int mhalf = (lane >> 4) << 3;
#pragma unroll
  for (int r = 0; r < 8; ++r) {
    const size_t idx = (size_t)(mtile * 16 + r + mhalf) * Dz + n;
    qo[idx] = (bf16_t)(aq[r] + bq);
    ko[idx] = (bf16_t)(ak[r] + bk);
    vo[idx] = (bf16_t)(av[r] + bv);
  }
}

// ---------------------------------------------------------------------------
// WMMA GEMM, 4 N-tiles per wave (A fragment reused 4x from registers):
// out[M,N] = A[M,K](bf16) @ Wt[N,K]^T + bias.  EPI: 2 -> f32 + add1.
// grid = (N/64, M/64), block = 128.
// ---------------------------------------------------------------------------
template <int EPI>
__global__ void gemm_kernel(const bf16_t* __restrict__ A, const bf16_t* __restrict__ Wt,
                            const float* __restrict__ bias,
                            float* __restrict__ outf,
                            const float* __restrict__ add1, int N, int K) {
  const int wave = threadIdx.x >> 5, lane = threadIdx.x & 31;
  const int ntile0 = blockIdx.x * 4;
  const int mtile = blockIdx.y * 4 + wave;
  const bf16_t* Arow = A + (size_t)mtile * 16 * K;
  v8f acc[4];
#pragma unroll
  for (int t = 0; t < 4; ++t) acc[t] = zero8();
  for (int k0 = 0; k0 < K; k0 += 32) {
    const v16bf a = load_a16(Arow, K, k0, lane);
#pragma unroll
    for (int t = 0; t < 4; ++t)
      acc[t] = wmma_bf16(a, load_b16(Wt + (size_t)(ntile0 + t) * 16 * K, K, k0, lane),
                         acc[t]);
  }
  const int mhalf = (lane >> 4) << 3;
#pragma unroll
  for (int t = 0; t < 4; ++t) {
    const int n = (ntile0 + t) * 16 + (lane & 15);
    const float bv = bias[n];
#pragma unroll
    for (int r = 0; r < 8; ++r) {
      const size_t idx = (size_t)(mtile * 16 + r + mhalf) * N + n;
      float v = acc[t][r] + bv;
      if constexpr (EPI >= 2) v += add1[idx];
      outf[idx] = v;
    }
  }
}

// ---------------------------------------------------------------------------
// Down-projection GEMM with TDM-staged A tiles:
//   h = act @ mWt^T + mb + h + xn        (reference: h = xn + res + mlp)
// A chunks (64 rows x 64 K) are DMA'd into LDS by wave 0 via tensor_load_to_lds,
// double buffered; TENSORcnt + workgroup barrier synchronize. 4 N-tiles/wave.
// grid = (D/64, MTOT/64), block = 128.
// ---------------------------------------------------------------------------
__global__ void mlp_down_tdm_kernel(const bf16_t* __restrict__ A,   // [MTOT][HID]
                                    const bf16_t* __restrict__ Wt,  // [D][HID]
                                    const float* __restrict__ bias,
                                    float* __restrict__ h,
                                    const float* __restrict__ xn) {
  __shared__ bf16_t asA[2][64 * LDSLD];
  const int wave = threadIdx.x >> 5, lane = threadIdx.x & 31;
  const int ntile0 = blockIdx.x * 4;
  const int mblk = blockIdx.y;
  const int mtile = mblk * 4 + wave;
  const unsigned long long abase =
      (unsigned long long)(size_t)(const void*)(A + (size_t)mblk * 64 * HIDz);
  const unsigned lds0 = (unsigned)(size_t)(void*)&asA[0][0];
  const unsigned lds1 = (unsigned)(size_t)(void*)&asA[1][0];
  const int NCH = HIDz / TDK;  // 32 chunks

  if (wave == 0)
    tdm_load_tile(lds0, abase, TDK, 64, HIDz);

  v8f acc[4];
#pragma unroll
  for (int t = 0; t < 4; ++t) acc[t] = zero8();

  for (int c = 0; c < NCH; ++c) {
    if (wave == 0) {
      if (c + 1 < NCH) {
        tdm_load_tile((c & 1) ? lds0 : lds1,
                      abase + (unsigned long long)(c + 1) * TDK * sizeof(bf16_t),
                      TDK, 64, HIDz);
        __builtin_amdgcn_s_wait_tensorcnt(1);
      } else {
        __builtin_amdgcn_s_wait_tensorcnt(0);
      }
    }
    __syncthreads();
    const bf16_t* Ach = &asA[c & 1][(size_t)wave * 16 * LDSLD];
#pragma unroll
    for (int k0 = 0; k0 < TDK; k0 += 32) {
      const v16bf a = load_a16(Ach, LDSLD, k0, lane);
#pragma unroll
      for (int t = 0; t < 4; ++t)
        acc[t] = wmma_bf16(
            a, load_b16(Wt + (size_t)(ntile0 + t) * 16 * HIDz, HIDz, c * TDK + k0, lane),
            acc[t]);
    }
    __syncthreads();
  }

  const int mhalf = (lane >> 4) << 3;
#pragma unroll
  for (int t = 0; t < 4; ++t) {
    const int n = (ntile0 + t) * 16 + (lane & 15);
    const float bv = bias[n];
#pragma unroll
    for (int r = 0; r < 8; ++r) {
      const size_t idx = (size_t)(mtile * 16 + r + mhalf) * Dz + n;
      h[idx] = acc[t][r] + bv + h[idx] + xn[idx];
    }
  }
}

// ---------------------------------------------------------------------------
// Fused gated MLP up-projection: act = (y@gW_u + b_u) * gelu(y@gW_g + b_g)
// ---------------------------------------------------------------------------
__global__ void gated_mlp_kernel(const bf16_t* __restrict__ A,
                                 const bf16_t* __restrict__ gWt,
                                 const float* __restrict__ gb,
                                 bf16_t* __restrict__ act) {
  const int wave = threadIdx.x >> 5, lane = threadIdx.x & 31;
  const int ntile = blockIdx.x;
  const int mtile = blockIdx.y * 4 + wave;
  const bf16_t* Arow = A + (size_t)mtile * 16 * Dz;
  const bf16_t* Bu = gWt + (size_t)(ntile * 16) * Dz;
  const bf16_t* Bg = gWt + (size_t)(HIDz + ntile * 16) * Dz;
  v8f au = zero8(), ag = zero8();
  for (int k0 = 0; k0 < Dz; k0 += 32) {
    const v16bf a = load_a16(Arow, Dz, k0, lane);
    au = wmma_bf16(a, load_b16(Bu, Dz, k0, lane), au);
    ag = wmma_bf16(a, load_b16(Bg, Dz, k0, lane), ag);
  }
  const int n = ntile * 16 + (lane & 15);
  const float bu = gb[n], bg = gb[HIDz + n];
  const int mhalf = (lane >> 4) << 3;
#pragma unroll
  for (int r = 0; r < 8; ++r) {
    const int m = mtile * 16 + r + mhalf;
    const float u = au[r] + bu;
    const float g = ag[r] + bg;
    act[(size_t)m * HIDz + n] = (bf16_t)(u * gelu_f(g));
  }
}

// ---------------------------------------------------------------------------
// Attention: one wave per (batch, head, 16-query tile). QK^T and P@V on WMMA,
// scores (16 x SP f32, ~67 KB) in dynamic LDS (320 KB/WGP), wave-local softmax
// with ALiBi bias + key mask.
// ---------------------------------------------------------------------------
__device__ __forceinline__ v16bf load_p_frag(const float* __restrict__ sc,
                                             const float* __restrict__ invs,
                                             int c0, int lane) {
  const int m  = lane & 15;
  const int kb = (lane >> 4) << 4;
  const float is = invs[m];
  const float* p = sc + (size_t)m * SPz + c0 + kb;
  v16bf a;
#pragma unroll
  for (int j = 0; j < 8; ++j) a[j] = (bf16_t)(p[j] * is);
#pragma unroll
  for (int j = 0; j < 8; ++j) a[8 + j] = (bf16_t)(p[8 + j] * is);
  return a;
}

__global__ void attn_kernel(const bf16_t* __restrict__ q, const bf16_t* __restrict__ k,
                            const bf16_t* __restrict__ v, const float* __restrict__ kmask,
                            bf16_t* __restrict__ attn) {
  extern __shared__ float smem[];
  float* sc   = smem;               // [16][SPz]
  float* invs = smem + 16 * SPz;    // [16]
  const int lane = threadIdx.x;
  const int qt = blockIdx.x, head = blockIdx.y, bb = blockIdx.z;

  const bf16_t* qbase = q + ((size_t)(bb * SPz + qt * 16) * Dz + head * HDz);
  const v16bf aq0 = load_a16(qbase, Dz, 0, lane);
  const v16bf aq1 = load_a16(qbase, Dz, 32, lane);

  const float slope = exp2f(-(float)(head + 1));   // 8/H == 1
  const float scale = 0.125f;                      // 1/sqrt(HD)
  const int n = lane & 15;
  const int mhalf = (lane >> 4) << 3;

  for (int kt = 0; kt < SPz / 16; ++kt) {
    const bf16_t* kb = k + ((size_t)(bb * SPz + kt * 16) * Dz + head * HDz);
    v8f s8 = zero8();
    s8 = wmma_bf16(aq0, load_b16(kb, Dz, 0, lane), s8);
    s8 = wmma_bf16(aq1, load_b16(kb, Dz, 32, lane), s8);
    const int j = kt * 16 + n;
    const float km = kmask[bb * SPz + j];
#pragma unroll
    for (int r = 0; r < 8; ++r) {
      const int m = r + mhalf;
      const int i = qt * 16 + m;
      sc[(size_t)m * SPz + j] = s8[r] * scale - slope * fabsf((float)(i - j)) + km;
    }
  }

  for (int m = 0; m < 16; ++m) {
    float* row = sc + (size_t)m * SPz;
    float mx = -3.4e38f;
    for (int j = lane; j < SPz; j += 32) mx = fmaxf(mx, row[j]);
#pragma unroll
    for (int off = 16; off; off >>= 1) mx = fmaxf(mx, __shfl_xor(mx, off, 32));
    float sm = 0.f;
    for (int j = lane; j < SPz; j += 32) {
      const float e = __expf(row[j] - mx);
      row[j] = e;
      sm += e;
    }
#pragma unroll
    for (int off = 16; off; off >>= 1) sm += __shfl_xor(sm, off, 32);
    if (lane == 0) invs[m] = 1.f / sm;
  }

  v8f o0 = zero8(), o1 = zero8(), o2 = zero8(), o3 = zero8();
  for (int kt = 0; kt < SPz / 32; ++kt) {
    const v16bf ap = load_p_frag(sc, invs, kt * 32, lane);
    const bf16_t* vb = v + ((size_t)(bb * SPz + kt * 32) * Dz + head * HDz);
    o0 = wmma_bf16(ap, load_b16_strided(vb + 0,  Dz, lane), o0);
    o1 = wmma_bf16(ap, load_b16_strided(vb + 16, Dz, lane), o1);
    o2 = wmma_bf16(ap, load_b16_strided(vb + 32, Dz, lane), o2);
    o3 = wmma_bf16(ap, load_b16_strided(vb + 48, Dz, lane), o3);
  }
  bf16_t* obase = attn + ((size_t)(bb * SPz + qt * 16) * Dz + head * HDz);
#pragma unroll
  for (int r = 0; r < 8; ++r) {
    bf16_t* orow = obase + (size_t)(r + mhalf) * Dz;
    orow[0  + n] = (bf16_t)o0[r];
    orow[16 + n] = (bf16_t)o1[r];
    orow[32 + n] = (bf16_t)o2[r];
    orow[48 + n] = (bf16_t)o3[r];
  }
}

// ---------------------------------------------------------------------------
// Projection head: M=4 rows -> VALU; BN over the batch dim is register-local
// when one thread owns one output column.
// ---------------------------------------------------------------------------
__global__ void head_bn_kernel(const float* __restrict__ zin, const float* __restrict__ W,
                               const float* __restrict__ bias, const float* __restrict__ bng,
                               const float* __restrict__ bnb, float* __restrict__ zout,
                               int K, int N, int rowStrideIn) {
  const int n = blockIdx.x * blockDim.x + threadIdx.x;
  if (n >= N) return;
  float t0 = 0.f, t1 = 0.f, t2 = 0.f, t3 = 0.f;
  for (int kk = 0; kk < K; ++kk) {
    const float w = W[(size_t)kk * N + n];
    t0 += zin[(size_t)0 * rowStrideIn + kk] * w;
    t1 += zin[(size_t)1 * rowStrideIn + kk] * w;
    t2 += zin[(size_t)2 * rowStrideIn + kk] * w;
    t3 += zin[(size_t)3 * rowStrideIn + kk] * w;
  }
  const float bv = bias[n];
  t0 += bv; t1 += bv; t2 += bv; t3 += bv;
  const float mu = 0.25f * (t0 + t1 + t2 + t3);
  const float d0 = t0 - mu, d1 = t1 - mu, d2 = t2 - mu, d3 = t3 - mu;
  const float var = 0.25f * (d0 * d0 + d1 * d1 + d2 * d2 + d3 * d3);
  const float inv = rsqrtf(var + EPSz);
  const float g = bng[n], b = bnb[n];
  zout[(size_t)0 * N + n] = gelu_f(d0 * inv * g + b);
  zout[(size_t)1 * N + n] = gelu_f(d1 * inv * g + b);
  zout[(size_t)2 * N + n] = gelu_f(d2 * inv * g + b);
  zout[(size_t)3 * N + n] = gelu_f(d3 * inv * g + b);
}

__global__ void head_out_kernel(const float* __restrict__ z2, const float* __restrict__ W3,
                                const float* __restrict__ b3, float* __restrict__ out) {
  const int n = threadIdx.x;
  if (n >= PROJz) return;
#pragma unroll
  for (int r = 0; r < Bz; ++r) {
    float acc = b3[n];
    for (int kk = 0; kk < HIDz; ++kk)
      acc += z2[(size_t)r * HIDz + kk] * W3[(size_t)kk * PROJz + n];
    out[(size_t)r * PROJz + n] = acc;
  }
}

// ---------------------------------------------------------------------------
// Host-side orchestration (all launches on `stream`; graph-capture safe)
// ---------------------------------------------------------------------------
extern "C" void kernel_launch(void* const* d_in, const int* in_sizes, int n_in,
                              void* d_out, int out_size, void* d_ws, size_t ws_size,
                              hipStream_t stream) {
  (void)in_sizes; (void)n_in; (void)out_size; (void)ws_size;

  const int*   x     = (const int*)  d_in[0];
  const float* embed = (const float*)d_in[1];
  const float* cls   = (const float*)d_in[2];
  const float* ln1_g = (const float*)d_in[3];
  const float* ln1_b = (const float*)d_in[4];
  const float* qW    = (const float*)d_in[5];
  const float* qb    = (const float*)d_in[6];
  const float* kW    = (const float*)d_in[7];
  const float* kb    = (const float*)d_in[8];
  const float* vW    = (const float*)d_in[9];
  const float* vb    = (const float*)d_in[10];
  const float* oW    = (const float*)d_in[11];
  const float* ob    = (const float*)d_in[12];
  const float* ln2_g = (const float*)d_in[13];
  const float* ln2_b = (const float*)d_in[14];
  const float* gW    = (const float*)d_in[15];
  const float* gb    = (const float*)d_in[16];
  const float* mW    = (const float*)d_in[17];
  const float* mb    = (const float*)d_in[18];
  const float* pW1   = (const float*)d_in[19];
  const float* pb1   = (const float*)d_in[20];
  const float* bn1g  = (const float*)d_in[21];
  const float* bn1b  = (const float*)d_in[22];
  const float* pW2   = (const float*)d_in[23];
  const float* pb2   = (const float*)d_in[24];
  const float* bn2g  = (const float*)d_in[25];
  const float* bn2b  = (const float*)d_in[26];
  const float* pW3   = (const float*)d_in[27];
  const float* pb3   = (const float*)d_in[28];

  char* cur = (char*)d_ws;
  auto alloc = [&](size_t bytes) -> void* {
    void* p = (void*)cur;
    cur += (bytes + 255) & ~(size_t)255;
    return p;
  };
  const size_t DD = (size_t)Dz * Dz;
  bf16_t* qWt = (bf16_t*)alloc(sizeof(bf16_t) * DEPTHz * DD);
  bf16_t* kWt = (bf16_t*)alloc(sizeof(bf16_t) * DEPTHz * DD);
  bf16_t* vWt = (bf16_t*)alloc(sizeof(bf16_t) * DEPTHz * DD);
  bf16_t* oWt = (bf16_t*)alloc(sizeof(bf16_t) * DEPTHz * DD);
  bf16_t* gWt = (bf16_t*)alloc(sizeof(bf16_t) * DEPTHz * (size_t)2 * HIDz * Dz);
  bf16_t* mWt = (bf16_t*)alloc(sizeof(bf16_t) * DEPTHz * (size_t)Dz * HIDz);
  float*  h    = (float*) alloc(sizeof(float)  * (size_t)MTOT * Dz);
  float*  xnf  = (float*) alloc(sizeof(float)  * (size_t)MTOT * Dz);
  float*  yin  = (float*) alloc(sizeof(float)  * (size_t)MTOT * Dz);
  bf16_t* xnb  = (bf16_t*)alloc(sizeof(bf16_t) * (size_t)MTOT * Dz);
  bf16_t* qbuf = (bf16_t*)alloc(sizeof(bf16_t) * (size_t)MTOT * Dz);
  bf16_t* kbuf = (bf16_t*)alloc(sizeof(bf16_t) * (size_t)MTOT * Dz);
  bf16_t* vbuf = (bf16_t*)alloc(sizeof(bf16_t) * (size_t)MTOT * Dz);
  bf16_t* abuf = (bf16_t*)alloc(sizeof(bf16_t) * (size_t)MTOT * Dz);
  bf16_t* ybuf = (bf16_t*)alloc(sizeof(bf16_t) * (size_t)MTOT * Dz);
  bf16_t* act  = (bf16_t*)alloc(sizeof(bf16_t) * (size_t)MTOT * HIDz);
  float*  kmsk = (float*) alloc(sizeof(float)  * Bz * SPz);
  float*  z1   = (float*) alloc(sizeof(float)  * Bz * HIDz);
  float*  z2   = (float*) alloc(sizeof(float)  * Bz * HIDz);

  {
    const int ne = Dz * Dz;
    dim3 g((ne + 255) / 256, 1, DEPTHz);
    tcast_kernel<<<g, 256, 0, stream>>>(qW, qWt, Dz, Dz);
    tcast_kernel<<<g, 256, 0, stream>>>(kW, kWt, Dz, Dz);
    tcast_kernel<<<g, 256, 0, stream>>>(vW, vWt, Dz, Dz);
    tcast_kernel<<<g, 256, 0, stream>>>(oW, oWt, Dz, Dz);
    const int ng = Dz * 2 * HIDz;
    tcast_kernel<<<dim3((ng + 255) / 256, 1, DEPTHz), 256, 0, stream>>>(gW, gWt, Dz, 2 * HIDz);
    const int nm = HIDz * Dz;
    tcast_kernel<<<dim3((nm + 255) / 256, 1, DEPTHz), 256, 0, stream>>>(mW, mWt, HIDz, Dz);
  }

  embed_kernel<<<MTOT, 256, 0, stream>>>(x, embed, cls, h);
  mask_kernel<<<(Bz * SPz + 255) / 256, 256, 0, stream>>>(x, kmsk);

  const dim3 gQKV(Dz / 16, MTOT / 64);      // fused QKV (N-tile 16, 3 mats)
  const dim3 gN4(Dz / 64, MTOT / 64);       // N-blocked GEMMs (N-tile 64)
  const dim3 gH(HIDz / 16, MTOT / 64);      // gated up-proj
  const size_t attn_lds = (size_t)(16 * SPz + 16) * sizeof(float);

  for (int d = 0; d < DEPTHz; ++d) {
    const size_t wd = (size_t)d * DD;
    ln_kernel<<<MTOT, 256, 0, stream>>>(h, ln1_g + d * Dz, ln1_b + d * Dz, xnf, xnb);
    qkv_kernel<<<gQKV, 128, 0, stream>>>(xnb, qWt + wd, kWt + wd, vWt + wd,
                                         qb + d * Dz, kb + d * Dz, vb + d * Dz,
                                         qbuf, kbuf, vbuf);
    attn_kernel<<<dim3(SPz / 16, Hz, Bz), 32, attn_lds, stream>>>(qbuf, kbuf, vbuf,
                                                                  kmsk, abuf);
    // yin = attn @ oW + ob + h
    gemm_kernel<2><<<gN4, 128, 0, stream>>>(abuf, oWt + wd, ob + d * Dz, yin, h,
                                            Dz, Dz);
    ln_kernel<<<MTOT, 256, 0, stream>>>(yin, ln2_g + d * Dz, ln2_b + d * Dz,
                                        nullptr, ybuf);
    gated_mlp_kernel<<<gH, 128, 0, stream>>>(ybuf, gWt + (size_t)d * 2 * HIDz * Dz,
                                             gb + (size_t)d * 2 * HIDz, act);
    // h = act @ mW + mb + h + xn    (TDM-staged A, double buffered in LDS)
    mlp_down_tdm_kernel<<<gN4, 128, 0, stream>>>(act, mWt + (size_t)d * Dz * HIDz,
                                                 mb + d * Dz, h, xnf);
  }

  head_bn_kernel<<<(HIDz + 255) / 256, 256, 0, stream>>>(h, pW1, pb1, bn1g, bn1b, z1,
                                                         Dz, HIDz, SPz * Dz);
  head_bn_kernel<<<(HIDz + 255) / 256, 256, 0, stream>>>(z1, pW2, pb2, bn2g, bn2b, z2,
                                                         HIDz, HIDz, HIDz);
  head_out_kernel<<<1, 128, 0, stream>>>(z2, pW3, pb3, (float*)d_out);
}
